// MoELinear_36197984371520
// MI455X (gfx1250) — matrix operand order repaired and checked by reading, not compiled, not run on hardware
//
#include <hip/hip_runtime.h>
#include <hip/hip_bf16.h>

// -----------------------------------------------------------------------------
// Grouped (ragged) GEMM for MoE linear on gfx1250 (MI455X, wave32, WMMA).
//
//   y[t, :] = x[t, :] @ W[expert(t)] + bias
//   x: [T, 512] f32, W: [E, 512, 512] f32, bias: [512] f32, groups contiguous.
//
// Roofline: 34.4 GFLOP, 335 MB compulsory -> ~14.4 us HBM floor @ 23.3 TB/s.
// fp32-grade accuracy on the bf16 matrix pipe via split-product:
//   a = a_hi + a_lo (bf16), a*b ~= a_hi*b_hi + a_hi*b_lo + a_lo*b_hi
// (3x v_wmma_f32_16x16x32_bf16 per 16x16x32 step; ~3x cheaper than native
//  f32 WMMA's K=4 path, ~2^-16 relative error).
//
// Pipeline: double-buffered LDS (80 KB), global loads for chunk c+1 issued
// before computing chunk c, one barrier per iteration. N-tiles on the fast
// grid axis + non-temporal output stores keep weights/inputs L2-resident.
// -----------------------------------------------------------------------------

typedef __attribute__((ext_vector_type(16))) __bf16         v16bf;
typedef __attribute__((ext_vector_type(8)))  float          v8f;
typedef __attribute__((ext_vector_type(4)))  unsigned short u16x4;
typedef __attribute__((ext_vector_type(4)))  unsigned int   u32x4;

#define TILE_M 128
#define TILE_N 128
#define TILE_K 32
#define LDA    40   // LDS row stride (bf16 elems): 80 B rows -> 16B-aligned b128,
                    // start bank 5*m mod 16 (coprime) -> conflict-free half-wave
#define DIM    512  // D_IN == D_OUT

__device__ __forceinline__ unsigned short f32_to_bf16_rne(float f) {
    unsigned int u = __builtin_bit_cast(unsigned int, f);
    unsigned int r = u + 0x7FFFu + ((u >> 16) & 1u);
    return (unsigned short)(r >> 16);
}
__device__ __forceinline__ unsigned short f32_to_bf16_trunc(float f) {
    return (unsigned short)(__builtin_bit_cast(unsigned int, f) >> 16);
}
__device__ __forceinline__ float bf16_to_f32(unsigned short h) {
    unsigned int u = ((unsigned int)h) << 16;
    return __builtin_bit_cast(float, u);
}

__device__ __forceinline__ v16bf lds_frag16(const unsigned short* base, int off0, int off1) {
    union { v16bf v; u32x4 q[2]; } t;
    t.q[0] = *reinterpret_cast<const u32x4*>(base + off0);  // ds_load_b128
    t.q[1] = *reinterpret_cast<const u32x4*>(base + off1);  // ds_load_b128
    return t.v;
}

__device__ __forceinline__ void split4(const float4 v, u16x4& hi, u16x4& lo) {
    hi.x = f32_to_bf16_rne(v.x); lo.x = f32_to_bf16_trunc(v.x - bf16_to_f32(hi.x));
    hi.y = f32_to_bf16_rne(v.y); lo.y = f32_to_bf16_trunc(v.y - bf16_to_f32(hi.y));
    hi.z = f32_to_bf16_rne(v.z); lo.z = f32_to_bf16_trunc(v.z - bf16_to_f32(hi.z));
    hi.w = f32_to_bf16_rne(v.w); lo.w = f32_to_bf16_trunc(v.w - bf16_to_f32(hi.w));
}

__global__ __launch_bounds__(256)
void moe_grouped_gemm_bf16x3(const float* __restrict__ x,
                             const float* __restrict__ w,
                             const float* __restrict__ bias,
                             const int*   __restrict__ group_sizes,
                             float*       __restrict__ y,
                             int T, int E) {
    // Double-buffered hi/lo bf16 planes. A: [row][k], B transposed: [n][k].
    __shared__ unsigned short Ahi[2][TILE_M * LDA];
    __shared__ unsigned short Alo[2][TILE_M * LDA];
    __shared__ unsigned short Bhi[2][TILE_N * LDA];
    __shared__ unsigned short Blo[2][TILE_N * LDA];

    const int tid   = threadIdx.x;
    const int lane  = tid & 31;
    const int wave  = tid >> 5;
    const int waveM = wave & 3;        // 4 waves x 32 rows
    const int waveN = wave >> 2;       // 2 waves x 64 cols
    const int half  = lane >> 4;
    const int lrow  = lane & 15;

    const int n0   = blockIdx.x * TILE_N;   // N-tiles: fast axis (L2 input reuse)
    const int row0 = blockIdx.y * TILE_M;   // M-tiles: slow axis (weights stay hot)

    // ---- expert of this token tile (contiguous groups; tile-aligned here) ----
    int expert = E - 1;
    {
        int accum = 0;
        for (int e = 0; e < E; ++e) {
            int gs = group_sizes[e];
            if (row0 < accum + gs) { expert = e; break; }
            accum += gs;
        }
    }
    const float* wmat = w + (size_t)expert * DIM * DIM;

    // Per-thread staging indices (constant over chunks).
    // A: 1024 float4 slots, 8 per 32-elem row.
    int aIdx[4], aRow[4], aC4[4];
    // B: 1024 float4 slots, 32 per 128-elem row.
    int bKr[4], bN[4];
#pragma unroll
    for (int i = 0; i < 4; ++i) {
        int idx = tid + i * 256;
        aRow[i] = idx >> 3;
        aC4[i]  = (idx & 7) * 4;
        int grow = row0 + aRow[i]; if (grow >= T) grow = T - 1;
        aIdx[i] = grow;
        bKr[i]  = idx >> 5;
        bN[i]   = (idx & 31) * 4;
    }

    v8f acc[2][4];
#pragma unroll
    for (int mi = 0; mi < 2; ++mi)
#pragma unroll
        for (int ni = 0; ni < 4; ++ni) acc[mi][ni] = {};

    const int nChunks = DIM / TILE_K;   // 16
    float4 va[4], vb[4];

    // ---- prologue: load + convert + store chunk 0 into buffer 0 ----
#pragma unroll
    for (int i = 0; i < 4; ++i) {
        va[i] = *reinterpret_cast<const float4*>(x + (size_t)aIdx[i] * DIM + aC4[i]);
        vb[i] = *reinterpret_cast<const float4*>(wmat + (size_t)bKr[i] * DIM + n0 + bN[i]);
    }
#pragma unroll
    for (int i = 0; i < 4; ++i) {
        u16x4 hi, lo;
        split4(va[i], hi, lo);
        *reinterpret_cast<u16x4*>(&Ahi[0][aRow[i] * LDA + aC4[i]]) = hi;
        *reinterpret_cast<u16x4*>(&Alo[0][aRow[i] * LDA + aC4[i]]) = lo;
        float f[4] = {vb[i].x, vb[i].y, vb[i].z, vb[i].w};
#pragma unroll
        for (int j = 0; j < 4; ++j) {
            Bhi[0][(bN[i] + j) * LDA + bKr[i]] = f32_to_bf16_rne(f[j]);
            Blo[0][(bN[i] + j) * LDA + bKr[i]] =
                f32_to_bf16_trunc(f[j] - bf16_to_f32(f32_to_bf16_rne(f[j])));
        }
    }
    __syncthreads();

    for (int c = 0; c < nChunks; ++c) {
        const int  cur     = c & 1;
        const bool hasNext = (c + 1) < nChunks;
        const int  kn      = (c + 1) * TILE_K;

        // ---- issue next chunk's global loads (overlap with WMMA below) ----
        if (hasNext) {
#pragma unroll
            for (int i = 0; i < 4; ++i) {
                const float4* pa =
                    reinterpret_cast<const float4*>(x + (size_t)aIdx[i] * DIM + kn + aC4[i]);
                const float4* pb =
                    reinterpret_cast<const float4*>(wmat + (size_t)(kn + bKr[i]) * DIM + n0 + bN[i]);
                va[i] = *pa;
                vb[i] = *pb;
                if (c + 2 < nChunks) {   // global_prefetch_b8 for chunk c+2
                    __builtin_prefetch((const char*)pa + TILE_K * sizeof(float), 0, 1);
                    __builtin_prefetch((const char*)pb + TILE_K * DIM * sizeof(float), 0, 1);
                }
            }
        }

        // ---- compute chunk c from LDS buffer `cur` ----
        const unsigned short* cAhi = Ahi[cur];
        const unsigned short* cAlo = Alo[cur];
        const unsigned short* cBhi = Bhi[cur];
        const unsigned short* cBlo = Blo[cur];

        // A 16x32 bf16 frag: lanes 0-15 M=0..15 {V0-3:K0-7, V4-7:K16-23},
        //                    lanes 16-31 same M  {K8-15,  K24-31}.
        v16bf aHi[2], aLo[2];
#pragma unroll
        for (int mi = 0; mi < 2; ++mi) {
            int rb   = waveM * 32 + mi * 16 + lrow;
            int off0 = rb * LDA + half * 8;
            int off1 = rb * LDA + 16 + half * 8;
            aHi[mi] = lds_frag16(cAhi, off0, off1);
            aLo[mi] = lds_frag16(cAlo, off0, off1);
        }
        // B 32x16 frag from Bt[n][k]: lanes 0-15 {K0-7, K16-23}, lanes 16-31 {K8-15, K24-31}.
#pragma unroll
        for (int ni = 0; ni < 4; ++ni) {
            int cb   = waveN * 64 + ni * 16 + lrow;
            int off0 = cb * LDA + half * 8;
            v16bf bH = lds_frag16(cBhi, off0, off0 + 16);
            v16bf bL = lds_frag16(cBlo, off0, off0 + 16);
#pragma unroll
            for (int mi = 0; mi < 2; ++mi) {
                acc[mi][ni] = __builtin_amdgcn_wmma_f32_16x16x32_bf16(
                    false, aHi[mi], false, bH, (short)0, acc[mi][ni], false, false);
                acc[mi][ni] = __builtin_amdgcn_wmma_f32_16x16x32_bf16(
                    false, aHi[mi], false, bL, (short)0, acc[mi][ni], false, false);
                acc[mi][ni] = __builtin_amdgcn_wmma_f32_16x16x32_bf16(
                    false, aLo[mi], false, bH, (short)0, acc[mi][ni], false, false);
            }
        }

        // ---- convert + store next chunk into the other buffer ----
        if (hasNext) {
            const int nxt = 1 - cur;
#pragma unroll
            for (int i = 0; i < 4; ++i) {
                u16x4 hi, lo;
                split4(va[i], hi, lo);
                *reinterpret_cast<u16x4*>(&Ahi[nxt][aRow[i] * LDA + aC4[i]]) = hi;
                *reinterpret_cast<u16x4*>(&Alo[nxt][aRow[i] * LDA + aC4[i]]) = lo;
                u16x4 bh, bl;
                split4(vb[i], bh, bl);
                unsigned short hs[4] = {bh.x, bh.y, bh.z, bh.w};
                unsigned short ls[4] = {bl.x, bl.y, bl.z, bl.w};
#pragma unroll
                for (int j = 0; j < 4; ++j) {
                    Bhi[nxt][(bN[i] + j) * LDA + bKr[i]] = hs[j];
                    Blo[nxt][(bN[i] + j) * LDA + bKr[i]] = ls[j];
                }
            }
            __syncthreads();   // single barrier per iteration (see pipeline note)
        }
    }

    // ---- epilogue: +bias, non-temporal f32 stores (write-once stream) ----
    // C/D 16x16 f32 layout: VGPR j -> M = j + 8*(lane>>4), lanes 0-15 N=0..15.
    const int nlane = lane & 15;
    const int mhalf = lane >> 4;
#pragma unroll
    for (int ni = 0; ni < 4; ++ni) {
        int col = n0 + waveN * 64 + ni * 16 + nlane;
        float bv = bias[col];
#pragma unroll
        for (int mi = 0; mi < 2; ++mi) {
            int rtop = row0 + waveM * 32 + mi * 16;
#pragma unroll
            for (int j = 0; j < 8; ++j) {
                int r = rtop + j + 8 * mhalf;
                if (r < T)
                    __builtin_nontemporal_store(acc[mi][ni][j] + bv,
                                                &y[(size_t)r * DIM + col]);
            }
        }
    }
}

extern "C" void kernel_launch(void* const* d_in, const int* in_sizes, int n_in,
                              void* d_out, int out_size, void* d_ws, size_t ws_size,
                              hipStream_t stream) {
    const float* x    = (const float*)d_in[0];
    const float* w    = (const float*)d_in[1];
    const float* bias = (const float*)d_in[2];
    const int*   gsz  = (const int*)d_in[3];

    int T = in_sizes[0] / DIM;   // tokens
    int E = in_sizes[3];         // experts

    dim3 grid(DIM / TILE_N, (T + TILE_M - 1) / TILE_M);  // N fast, M slow
    dim3 block(256);
    moe_grouped_gemm_bf16x3<<<grid, block, 0, stream>>>(
        x, w, bias, gsz, (float*)d_out, T, E);
}
